// CNODExtmod_31696858644721
// MI455X (gfx1250) — matrix operand order repaired and checked by reading, not compiled, not run on hardware
//
#include <hip/hip_runtime.h>

// ---- problem constants ----
#define ID_   16
#define NC_   32
#define HID_  512
#define D0_   512
#define DIN_  528
#define KP1   544          // K of layer1 padded to 32
#define BSZ   256
#define T_    50
#define NSTEP 5
#define DT_   0.05f
#define HW    1040         // 2*D0 + ID (state width)

// output layout (f32, concatenated in return order)
#define OUT_PRED  0
#define OUT_TRAJ  204800
#define OUT_TIMES 409600
#define OUT_LASTH 409650
#define OUT_HFIN  540722

typedef __attribute__((ext_vector_type(16))) _Float16     v16h;
typedef __attribute__((ext_vector_type(8)))  float        v8f;
typedef __attribute__((ext_vector_type(2)))  float        v2f;
typedef __attribute__((ext_vector_type(4)))  unsigned int u32x4;

#define GAS __attribute__((address_space(1)))   // global address space

union Frag16 { v16h h16; u32x4 u4[2]; };

// Defeat LICM: make the weight base pointer opaque at each use site so the
// compiler cannot hoist the (loop-invariant) weight loads out of the time
// loops and spill them to scratch. Zero-cost empty asm.
__device__ __forceinline__ const _Float16* opaque_ptr(const _Float16* p) {
  asm volatile("" : "+s"(p));
  return p;
}

__device__ __forceinline__ v8f wmma16(const Frag16& a, const Frag16& b, v8f c) {
  return __builtin_amdgcn_wmma_f32_16x16x32_f16(false, a.h16, false, b.h16,
                                                (short)0, c, false, false);
}

// A fragment (16x32 f16, M = lane&15) from LDS:
// lanes<16 hold K {0..7,16..23}, lanes>=16 hold K {8..15,24..31}
__device__ __forceinline__ Frag16 ldAfrag(const _Float16* row, int kb, int hh) {
  Frag16 f;
  f.u4[0] = *(const u32x4*)(row + kb + hh * 8);
  f.u4[1] = *(const u32x4*)(row + kb + 16 + hh * 8);
  return f;
}
// B fragment (32x16 f16, N = lane&15) from GLOBAL memory (forced addrspace(1)
// so the opaque pointer still lowers to global_load_b128, not flat_load):
// lanes<16 hold K {0..7,16..23}, lanes>=16 hold K {8..15,24..31}
__device__ __forceinline__ Frag16 ldBfrag(const _Float16* wrow, int kb, int hh) {
  const GAS u32x4* p = (const GAS u32x4*)(uintptr_t)(wrow + kb + hh * 8);
  Frag16 f;
  f.u4[0] = p[0];
  f.u4[1] = p[2];
  return f;
}

// One GEMM layer: 16 x (nKt*32) @ (nKt*32) x (>=512), 4 N-tiles per wave (nt = wave + 8j)
__device__ __forceinline__ void gemm_block(
    const _Float16* act, int aStr, int nKt,
    const _Float16* Wt0, int kPad,
    const float* bias, bool relu,
    _Float16* outH, float* outF, int oStr,
    int wave, int lane)
{
  const _Float16* Wt = opaque_ptr(Wt0);      // block cross-loop hoisting of weight loads
  const int lh = lane & 15, hh = lane >> 4;
  const _Float16* aRow = act + lh * aStr;
  v8f acc[4] = {};
  const _Float16* wr[4];
  int n0[4];
#pragma unroll
  for (int j = 0; j < 4; ++j) {
    int nt = wave + 8 * j;
    n0[j] = nt * 16 + lh;
    wr[j] = Wt + (size_t)n0[j] * kPad;
  }
#pragma unroll 2
  for (int kt = 0; kt < nKt; ++kt) {
    int kb = kt * 32;
    Frag16 a = ldAfrag(aRow, kb, hh);
#pragma unroll
    for (int j = 0; j < 4; ++j) {
      Frag16 b = ldBfrag(wr[j], kb, hh);
      acc[j] = wmma16(a, b, acc[j]);
    }
  }
#pragma unroll
  for (int j = 0; j < 4; ++j) {
    float bs = bias[n0[j]];
#pragma unroll
    for (int r = 0; r < 8; ++r) {
      int m = r + 8 * hh;                    // C layout: M = r + 8*(lane>=16)
      float v = acc[j][r] + bs;
      if (relu) v = fmaxf(v, 0.f);
      if (outH) outH[m * oStr + n0[j]] = (_Float16)v;
      else      outF[m * oStr + n0[j]] = v;
    }
  }
}

// ---- prep: transpose+convert weights to f16 [N][Kpad] in workspace; copy times out ----
__global__ void cnode_prep(const float* __restrict__ W1, const float* __restrict__ W2,
                           const float* __restrict__ W3, const float* __restrict__ times,
                           _Float16* __restrict__ wsW, float* __restrict__ out)
{
  int stride = gridDim.x * blockDim.x;
  int gid = blockIdx.x * blockDim.x + threadIdx.x;
  _Float16* W1t = wsW;                       // 512 x 544
  _Float16* W2t = wsW + 512 * KP1;           // 512 x 512
  _Float16* W3t = W2t + 512 * 512;           // 528 x 512
  for (int i = gid; i < 512 * KP1; i += stride) {
    int n = i / KP1, k = i % KP1;
    float v = (k < DIN_) ? W1[(size_t)k * HID_ + n] : 0.f;
    W1t[i] = (_Float16)v;
  }
  for (int i = gid; i < 512 * 512; i += stride) {
    int n = i >> 9, k = i & 511;
    W2t[i] = (_Float16)W2[(size_t)k * HID_ + n];
  }
  for (int i = gid; i < DIN_ * 512; i += stride) {
    int n = i >> 9, k = i & 511;
    W3t[i] = (_Float16)W3[(size_t)k * DIN_ + n];
  }
  if (gid < T_) out[OUT_TIMES + gid] = times[gid];
}

// ---- resolve jnp.any(mask[:,t]) into last time index with any observation ----
__global__ void cnode_mask(const float* __restrict__ maskp, int* __restrict__ tLast)
{
  __shared__ int flags[64];
  int t = threadIdx.x;
  flags[t] = 0;
  __syncthreads();
  if (t < T_) {
    int f = 0;
    for (int b = 0; b < BSZ; ++b) f |= (maskp[(size_t)b * T_ + t] > 0.f) ? 1 : 0;
    flags[t] = f;
  }
  __syncthreads();
  if (t == 0) {
    int tl = -1;
    for (int q = 0; q < T_; ++q) if (flags[q]) tl = q;
    *tLast = tl;
  }
}

// ---- main persistent kernel: one block = 16 batch rows, 8 wave32s ----
__global__ __launch_bounds__(256)
void cnode_main(const float* __restrict__ Y, const float* __restrict__ maskp,
                const float* __restrict__ Aq, const float* __restrict__ Bvq,
                const float* __restrict__ b1, const float* __restrict__ b2,
                const float* __restrict__ b3,
                const _Float16* __restrict__ W1t, const _Float16* __restrict__ W2t,
                const _Float16* __restrict__ W3t, const int* __restrict__ tLastP,
                float* __restrict__ out)
{
  extern __shared__ char smem[];
  float*    hL   = (float*)(smem);               // 16 x 1040 f32 state     (66560 B)
  _Float16* bufA = (_Float16*)(smem + 66560);    // act1 16x544 / act3 16x512 (17408 B)
  _Float16* bufB = (_Float16*)(smem + 83968);    // act2 16x512             (16384 B)
  float*    drv  = (float*)(smem + 100352);      // driver 16x528 f32       (33792 B)
  float*    b1s  = (float*)(smem + 134144);      // 512
  float*    b2s  = (float*)(smem + 136192);      // 512
  float*    b3s  = (float*)(smem + 138240);      // 528
  float*    AL   = (float*)(smem + 140352);      // HiPPO A 32x32 f32
  float*    BvL  = (float*)(smem + 144448);      // 32

  const int tid  = threadIdx.x;
  const int wave = tid >> 5, lane = tid & 31;
  const int lh = lane & 15, hh = lane >> 4;
  const int b0 = blockIdx.x * 16;

  for (int e = tid; e < 16 * HW; e += 256) hL[e] = 0.f;
  for (int e = tid; e < 512; e += 256) { b1s[e] = b1[e]; b2s[e] = b2[e]; }
  for (int e = tid; e < DIN_; e += 256) b3s[e] = b3[e];
  for (int e = tid; e < 1024; e += 256) AL[e] = Aq[e];
  if (tid < 32) BvL[tid] = Bvq[tid];
  __syncthreads();

  const float bv0 = BvL[lh], bv1 = BvL[16 + lh];
  const int tl = *tLastP;

  for (int t = 0; t < T_; ++t) {
    for (int s = 0; s < NSTEP; ++s) {
      // ---- build f16 activation tile act1 = cn1 (K padded to 544) ----
      for (int e = tid; e < 16 * KP1; e += 256) {
        int m = e / KP1, k = e % KP1;
        float v = (k < DIN_) ? hL[m * HW + D0_ + k] : 0.f;
        bufA[e] = (_Float16)v;
      }
      __syncthreads();
      // ---- MLP layer 1: (16,544)@(544,512), relu ----
      gemm_block(bufA, KP1, 17, W1t, KP1, b1s, true, bufB, nullptr, HID_, wave, lane);
      __syncthreads();
      // ---- MLP layer 2: (16,512)@(512,512), relu (output aliases act1) ----
      gemm_block(bufB, HID_, 16, W2t, HID_, b2s, true, bufA, nullptr, HID_, wave, lane);
      __syncthreads();
      // ---- MLP layer 3: (16,512)@(512,528) -> driver f32; 33rd N-tile on wave 0 ----
      gemm_block(bufA, HID_, 16, W3t, HID_, b3s, false, nullptr, drv, DIN_, wave, lane);
      if (wave == 0) {
        const _Float16* aRow = bufA + lh * HID_;
        int n = 32 * 16 + lh;
        const _Float16* wrow = opaque_ptr(W3t) + (size_t)n * HID_;
        v8f acc = {};
#pragma unroll 2
        for (int kt = 0; kt < 16; ++kt) {
          Frag16 a = ldAfrag(aRow, kt * 32, hh);
          Frag16 b = ldBfrag(wrow, kt * 32, hh);
          acc = wmma16(a, b, acc);
        }
        float bs = b3s[n];
#pragma unroll
        for (int r = 0; r < 8; ++r) drv[(r + 8 * hh) * DIN_ + n] = acc[r] + bs;
      }
      // ---- HiPPO einsum in full f32 WMMA: rows 2*wave, 2*wave+1 ----
      v8f e00 = {}, e01 = {}, e10 = {}, e11 = {};
      {
        const float* hr0 = hL + (2 * wave) * HW;
        const float* hr1 = hL + (2 * wave + 1) * HW;
#pragma unroll
        for (int kt = 0; kt < 8; ++kt) {
          int k = kt * 4 + hh * 2;
          // B[k][n] = A[n][k]  (einsum 'bik,nk->bin')
          v2f bA = *(const v2f*)(AL + lh * 32 + k);
          v2f bB = *(const v2f*)(AL + (16 + lh) * 32 + k);
          v2f a0 = *(const v2f*)(hr0 + lh * 32 + k);
          v2f a1 = *(const v2f*)(hr1 + lh * 32 + k);
          e00 = __builtin_amdgcn_wmma_f32_16x16x4_f32(false, a0, false, bA, (short)0, e00, false, false);
          e01 = __builtin_amdgcn_wmma_f32_16x16x4_f32(false, a0, false, bB, (short)0, e01, false, false);
          e10 = __builtin_amdgcn_wmma_f32_16x16x4_f32(false, a1, false, bA, (short)0, e10, false, false);
          e11 = __builtin_amdgcn_wmma_f32_16x16x4_f32(false, a1, false, bB, (short)0, e11, false, false);
        }
        // Bv[n] * cn1[m] term (read state BEFORE barrier to avoid racing the update)
#pragma unroll
        for (int r = 0; r < 8; ++r) {
          float c0 = hr0[D0_ + 8 * hh + r];
          float c1 = hr1[D0_ + 8 * hh + r];
          e00[r] += bv0 * c0; e01[r] += bv1 * c0;
          e10[r] += bv0 * c1; e11[r] += bv1 * c1;
        }
      }
      __syncthreads();
      // ---- Euler update: h[:, :512] += DT*cn_ode ; h[:, 512:] += DT*driver ----
      {
        float* hr0 = hL + (2 * wave) * HW;
        float* hr1 = hL + (2 * wave + 1) * HW;
#pragma unroll
        for (int r = 0; r < 8; ++r) {
          int m = r + 8 * hh;
          hr0[m * 32 + lh]      += DT_ * e00[r];
          hr0[m * 32 + 16 + lh] += DT_ * e01[r];
          hr1[m * 32 + lh]      += DT_ * e10[r];
          hr1[m * 32 + 16 + lh] += DT_ * e11[r];
        }
      }
      for (int e = tid; e < 16 * DIN_; e += 256) {
        int m = e / DIN_, c = e % DIN_;
        hL[m * HW + D0_ + c] += DT_ * drv[e];
      }
      __syncthreads();
    } // euler steps

    // ---- observation: pred = h_end[:, 512:528] ----
    {
      int rb = tid >> 4, ci = tid & 15;
      float pv = hL[rb * HW + D0_ + ci];
      size_t po = (size_t)(b0 + rb) * (T_ * ID_) + (size_t)t * ID_ + ci;
      out[OUT_PRED + po] = pv;
      out[OUT_TRAJ + po] = pv;
    }
    __syncthreads();
    // ---- masked state update: h_upd = [h_cn, y_t, h_cn]; blend with mask ----
    for (int e = tid; e < 16 * DIN_; e += 256) {
      int rb = e / DIN_, c = e % DIN_;
      float m = maskp[(size_t)(b0 + rb) * T_ + t];
      float oldv = hL[rb * HW + D0_ + c];
      float newv = (c < ID_) ? Y[(size_t)(b0 + rb) * (T_ * ID_) + (size_t)t * ID_ + c]
                             : hL[rb * HW + (c - ID_)];
      hL[rb * HW + D0_ + c] = m * newv + (1.f - m) * oldv;
    }
    __syncthreads();
    if (t == tl) {
      for (int e = tid; e < 16 * D0_; e += 256) {
        int rb = e / D0_, c = e % D0_;
        out[OUT_LASTH + (size_t)(b0 + rb) * D0_ + c] = hL[rb * HW + c];
      }
    }
    if (t == T_ - 1) {
      for (int e = tid; e < 16 * HW; e += 256) {
        int rb = e / HW, c = e % HW;
        out[OUT_HFIN + (size_t)(b0 + rb) * HW + c] = hL[rb * HW + c];
      }
      if (tl < 0) {  // no observation at all -> last_h stays h0 = 0
        for (int e = tid; e < 16 * D0_; e += 256) {
          int rb = e / D0_, c = e % D0_;
          out[OUT_LASTH + (size_t)(b0 + rb) * D0_ + c] = 0.f;
        }
      }
    }
  }
}

extern "C" void kernel_launch(void* const* d_in, const int* in_sizes, int n_in,
                              void* d_out, int out_size, void* d_ws, size_t ws_size,
                              hipStream_t stream)
{
  (void)in_sizes; (void)n_in; (void)out_size; (void)ws_size;
  const float* times = (const float*)d_in[0];
  const float* Y     = (const float*)d_in[1];
  const float* maskp = (const float*)d_in[2];
  const float* Aq    = (const float*)d_in[3];
  const float* Bvq   = (const float*)d_in[4];
  const float* W1    = (const float*)d_in[5];
  const float* b1    = (const float*)d_in[6];
  const float* W2    = (const float*)d_in[7];
  const float* b2    = (const float*)d_in[8];
  const float* W3    = (const float*)d_in[9];
  const float* b3    = (const float*)d_in[10];
  float* out = (float*)d_out;

  _Float16* wsW = (_Float16*)d_ws;
  _Float16* W1t = wsW;
  _Float16* W2t = W1t + 512 * KP1;
  _Float16* W3t = W2t + 512 * 512;
  int* tLast = (int*)(W3t + DIN_ * 512);

  cnode_prep<<<512, 256, 0, stream>>>(W1, W2, W3, times, wsW, out);
  cnode_mask<<<1, 64, 0, stream>>>(maskp, tLast);

  const size_t shmem = 144576;  // ~141 KB of the WGP's 320 KB LDS
  (void)hipFuncSetAttribute((const void*)cnode_main,
                            hipFuncAttributeMaxDynamicSharedMemorySize, (int)shmem);
  cnode_main<<<BSZ / 16, 256, shmem, stream>>>(Y, maskp, Aq, Bvq, b1, b2, b3,
                                               W1t, W2t, W3t, tLast, out);
}
